// SimpleLSTM_27650999452464
// MI455X (gfx1250) — matrix operand order repaired and checked by reading, not compiled, not run on hardware
//
#include <hip/hip_runtime.h>

// ---------------------------------------------------------------------------
// MI455X (gfx1250) 3-layer LSTM, bf16 WMMA (v_wmma_f32_16x16x32_bf16),
// fp32 accumulation, fp32 cell state. Persistent recurrent kernel with
// device-scope spin barrier (cluster barriers are NOP without cluster launch).
// GEMM inner loops are software-pipelined (ping-pong double buffering) so the
// matrix pipe overlaps the fragment loads; with only ~2 waves/SIMD in the
// recurrent kernel, latency hiding must come from the instruction stream.
// ---------------------------------------------------------------------------

typedef __attribute__((ext_vector_type(16))) __bf16 bf16x16;
typedef __attribute__((ext_vector_type(8)))  float  f32x8;
typedef __attribute__((ext_vector_type(4)))  int    i32x4;

struct Frag32 { i32x4 lo, hi; };   // 32 bytes == one v16bf fragment

#define NB     64                  // batch
#define LT     1024                // seq length
#define MROWS  (NB * LT)           // 65536 GEMM rows (n*L + t)
#define HD     512                 // hidden
#define G4     2048                // 4*H gate rows
#define FEAT   51
#define K1     128                 // layer-1 K (102 padded to 128)
#define RNN_WGS 8                  // persistent workgroups (64 h-cols each)

__device__ __forceinline__ unsigned short f2bf(float f) {
  unsigned u = __float_as_uint(f);
  u += 0x7FFFu + ((u >> 16) & 1u);          // round-to-nearest-even
  return (unsigned short)(u >> 16);
}

// A fragment 16x32 bf16 from row-major [.. x ld] source (ISA 16-bit A layout):
// lanes 0-15 : row=lane,    K = {0..7, 16..23}
// lanes 16-31: row=lane-16, K = {8..15, 24..31}
__device__ __forceinline__ bf16x16 load_a(const unsigned short* base, long ld, int lane) {
  const int half = lane >> 4;
  const unsigned short* p = base + (long)(lane & 15) * ld + 8 * half;
  Frag32 f;
  f.lo = *(const i32x4*)(p);        // K = 8h .. 8h+7
  f.hi = *(const i32x4*)(p + 16);   // K = 16+8h .. 16+8h+7
  return __builtin_bit_cast(bf16x16, f);
}

// B fragment 32x16: B = W^T with W row-major [n][k] (lane owns output col n):
// lanes 0-15: col=lane, K={0..7,16..23}; lanes 16-31: col=lane-16, K={8..15,24..31}
__device__ __forceinline__ bf16x16 load_b(const unsigned short* base, long ld, int lane) {
  const unsigned short* p = base + (long)(lane & 15) * ld + 8 * (lane >> 4);
  Frag32 f;
  f.lo = *(const i32x4*)(p);
  f.hi = *(const i32x4*)(p + 16);
  return __builtin_bit_cast(bf16x16, f);
}

__device__ __forceinline__ f32x8 wmma_bf16(bf16x16 a, bf16x16 b, f32x8 c) {
  return __builtin_amdgcn_wmma_f32_16x16x32_bf16(false, a, false, b, (short)0, c,
                                                 false, false);
}

__device__ __forceinline__ float sigm(float x) { return 1.0f / (1.0f + __expf(-x)); }

// fast tanh from __expf (one transcendental, one divide)
__device__ __forceinline__ float fast_tanh(float x) {
  const float e = __expf(-2.0f * fabsf(x));
  const float r = (1.0f - e) / (1.0f + e);
  return copysignf(r, x);
}

// ---------------------------------------------------------------------------
// Packing / small kernels
// ---------------------------------------------------------------------------
__global__ void k_pack_z0(const float* __restrict__ x, const float* __restrict__ w,
                          unsigned short* __restrict__ z) {
  long idx = (long)blockIdx.x * blockDim.x + threadIdx.x;   // < MROWS*K1
  int  col = (int)(idx & (K1 - 1));
  long row = idx >> 7;
  float v = 0.0f;
  if (col < FEAT)          v = x[row * FEAT + col];
  else if (col < 2 * FEAT) v = w[row * FEAT + (col - FEAT)];
  z[idx] = f2bf(v);
}

__global__ void k_pack_mat(const float* __restrict__ s, unsigned short* __restrict__ d,
                           int Rs, int Cs, int Cd, long total) {
  long idx = (long)blockIdx.x * blockDim.x + threadIdx.x;
  if (idx >= total) return;
  int  c = (int)(idx % Cd);
  long r = idx / Cd;
  float v = (r < Rs && c < Cs) ? s[r * Cs + c] : 0.0f;
  d[idx] = f2bf(v);
}

__global__ void k_bias_add(const float* __restrict__ a, const float* __restrict__ b,
                           float* __restrict__ o, int n) {
  int i = blockIdx.x * blockDim.x + threadIdx.x;
  if (i < n) o[i] = a[i] + b[i];
}

// ---------------------------------------------------------------------------
// Input projection GEMM: XP[M x 2048] = A[M x K](bf16) @ W[2048 x K]^T + bias
// grid = (N/128, M/64); block = 256 (8 waves). Wave w: m-tile = w>>1,
// n-tiles = (w&1)*4 + {0..3}. bx fastest => 16 N-blocks reuse the A tile in L2.
// Ping-pong software pipeline over K (KT is even: 4 or 16): no register
// rotation copies, loads for step kt+1 fly during step kt's WMMA chain.
// ---------------------------------------------------------------------------
__global__ __launch_bounds__(256) void k_proj_gemm(
    const unsigned short* __restrict__ A, const unsigned short* __restrict__ W,
    const float* __restrict__ bias, float* __restrict__ XP, int K) {
  const int lane = threadIdx.x & 31;
  const int wave = threadIdx.x >> 5;
  const long m0 = (long)blockIdx.y * 64 + (wave >> 1) * 16;
  const int  n0 = blockIdx.x * 128 + (wave & 1) * 64;

  const unsigned short* pA = A + m0 * K;
  const unsigned short* pW[4];
#pragma unroll
  for (int i = 0; i < 4; ++i) pW[i] = W + (long)(n0 + i * 16) * K;

  f32x8 acc[4] = {};
  const int KT = K >> 5;                       // 4 or 16 (even)

  bf16x16 a0 = load_a(pA, K, lane);
  bf16x16 b0[4];
#pragma unroll
  for (int i = 0; i < 4; ++i) b0[i] = load_b(pW[i], K, lane);
  bf16x16 a1;
  bf16x16 b1[4];

  for (int kt = 0; kt + 2 < KT; kt += 2) {
    const int kA = (kt + 1) << 5;
    a1 = load_a(pA + kA, K, lane);
#pragma unroll
    for (int i = 0; i < 4; ++i) b1[i] = load_b(pW[i] + kA, K, lane);
#pragma unroll
    for (int i = 0; i < 4; ++i) acc[i] = wmma_bf16(a0, b0[i], acc[i]);

    const int kB = (kt + 2) << 5;
    a0 = load_a(pA + kB, K, lane);
#pragma unroll
    for (int i = 0; i < 4; ++i) b0[i] = load_b(pW[i] + kB, K, lane);
#pragma unroll
    for (int i = 0; i < 4; ++i) acc[i] = wmma_bf16(a1, b1[i], acc[i]);
  }
  // epilogue: a0/b0 hold step KT-2; load KT-1, then drain both
  {
    const int kL = (KT - 1) << 5;
    a1 = load_a(pA + kL, K, lane);
#pragma unroll
    for (int i = 0; i < 4; ++i) b1[i] = load_b(pW[i] + kL, K, lane);
#pragma unroll
    for (int i = 0; i < 4; ++i) acc[i] = wmma_bf16(a0, b0[i], acc[i]);
#pragma unroll
    for (int i = 0; i < 4; ++i) acc[i] = wmma_bf16(a1, b1[i], acc[i]);
  }

  const int half = lane >> 4, ccol = lane & 15;
#pragma unroll
  for (int i = 0; i < 4; ++i) {
    const int col = n0 + i * 16 + ccol;
    const float bv = bias[col];
#pragma unroll
    for (int r = 0; r < 8; ++r)
      XP[(m0 + r + 8 * half) * (long)G4 + col] = acc[i][r] + bv;
  }
}

// ---------------------------------------------------------------------------
// Persistent recurrent kernel. WG wg owns h-columns [wg*64, wg*64+64).
// Per step: gates[64 x 256] = h_prev[64 x 512] @ Whh_slice^T  (WMMA, fp32 acc)
//   column q = g*64 + jj  <->  Whh row g*512 + j0 + jj   (g = gate i/f/g/o)
// then fp32 elementwise LSTM update (c lives in registers for all 1024 steps),
// h written back as bf16, then a device-wide arrive+spin barrier.
// ---------------------------------------------------------------------------
__global__ __launch_bounds__(256) void k_lstm_rec(
    const unsigned short* __restrict__ Whh,   // [2048 x 512] bf16
    const float* __restrict__ XP,             // [MROWS x 2048] x_proj + bias
    const unsigned short* __restrict__ H0,    // [64 x 512] bf16 zeros
    unsigned short* __restrict__ HB,          // [MROWS x 512] bf16 out
    unsigned int* __restrict__ bar) {
  __shared__ float gates[64 * 256];           // 64 KB

  const int tid  = threadIdx.x;
  const int lane = tid & 31;
  const int wave = tid >> 5;
  const int j0   = blockIdx.x * 64;

  // GEMM tiling: wave -> m-tiles {2p,2p+1} (p=wave>>2), gate gq=wave&3,
  // n-tiles gq*4 + {0..3}  (2m x 4n = 8 accumulators / wave)
  const int p  = wave >> 2;
  const int gq = wave & 3;

  // Whh row-block base for this wave (constant over t and k)
  const unsigned short* pW[4];
#pragma unroll
  for (int i = 0; i < 4; ++i)
    pW[i] = Whh + ((long)gq * HD + j0 + i * 16) * HD;

  // elementwise ownership: batch row em, h-cols jj = esub + 4k, k=0..15
  const int em   = tid >> 2;
  const int esub = tid & 3;

  float c[16];
#pragma unroll
  for (int i = 0; i < 16; ++i) c[i] = 0.0f;

  for (int t = 0; t < LT; ++t) {
    // -------- GEMM phase --------
    const unsigned short* hbase;
    long rstride;
    if (t == 0) { hbase = H0;                       rstride = HD; }
    else        { hbase = HB + (long)(t - 1) * HD;  rstride = (long)LT * HD; }
    const unsigned short* pA0 = hbase + (long)(2 * p) * 16 * rstride;
    const unsigned short* pA1 = hbase + (long)(2 * p + 1) * 16 * rstride;

    // hint the cache about this step's x_proj rows
    {
      const float* xpp = XP + ((long)em * LT + t) * G4 + j0 + esub * 16;
      __builtin_prefetch(xpp, 0, 1);
      __builtin_prefetch(xpp + 512, 0, 1);
      __builtin_prefetch(xpp + 1024, 0, 1);
      __builtin_prefetch(xpp + 1536, 0, 1);
    }

    f32x8 acc[8] = {};
    // software pipeline: fragments for k-step kt+1 load during kt's WMMA chain
    bf16x16 a0 = load_a(pA0, rstride, lane);
    bf16x16 a1 = load_a(pA1, rstride, lane);
    bf16x16 b[4];
#pragma unroll
    for (int i = 0; i < 4; ++i) b[i] = load_b(pW[i], HD, lane);

#pragma unroll 4
    for (int kt = 1; kt < 16; ++kt) {
      const int k0 = kt << 5;
      bf16x16 na0 = load_a(pA0 + k0, rstride, lane);
      bf16x16 na1 = load_a(pA1 + k0, rstride, lane);
      bf16x16 nb[4];
#pragma unroll
      for (int i = 0; i < 4; ++i) nb[i] = load_b(pW[i] + k0, HD, lane);
#pragma unroll
      for (int i = 0; i < 4; ++i) {
        acc[i]     = wmma_bf16(a0, b[i], acc[i]);
        acc[4 + i] = wmma_bf16(a1, b[i], acc[4 + i]);
      }
      a0 = na0;
      a1 = na1;
#pragma unroll
      for (int i = 0; i < 4; ++i) b[i] = nb[i];
    }
#pragma unroll
    for (int i = 0; i < 4; ++i) {
      acc[i]     = wmma_bf16(a0, b[i], acc[i]);
      acc[4 + i] = wmma_bf16(a1, b[i], acc[4 + i]);
    }

    // store C fragments to LDS: lane col = gq*64 + i*16 + ccol, row = mt*16+r+8h
    {
      const int half = lane >> 4, ccol = lane & 15;
#pragma unroll
      for (int mi = 0; mi < 2; ++mi) {
        const int mrow = (2 * p + mi) * 16 + 8 * half;
#pragma unroll
        for (int i = 0; i < 4; ++i) {
          const int q = gq * 64 + i * 16 + ccol;
#pragma unroll
          for (int r = 0; r < 8; ++r)
            gates[(mrow + r) * 256 + q] = acc[mi * 4 + i][r];
        }
      }
    }
    __syncthreads();

    // -------- elementwise LSTM update (fp32) --------
    {
      const long xrow = (long)em * LT + t;
      const float* xp = XP + xrow * G4;
#pragma unroll
      for (int k = 0; k < 16; ++k) {
        const int jj = esub + 4 * k;
        const float gi = gates[em * 256 +       jj] + xp[          j0 + jj];
        const float gf = gates[em * 256 +  64 + jj] + xp[ 512 +    j0 + jj];
        const float gg = gates[em * 256 + 128 + jj] + xp[1024 +    j0 + jj];
        const float go = gates[em * 256 + 192 + jj] + xp[1536 +    j0 + jj];
        const float iv = sigm(gi);
        const float fv = sigm(gf);
        const float gv = fast_tanh(gg);
        const float ov = sigm(go);
        const float cv = fv * c[k] + iv * gv;
        c[k] = cv;
        HB[xrow * HD + j0 + jj] = f2bf(ov * fast_tanh(cv));
      }
    }

    // -------- device-wide step barrier (monotonic counter) --------
    __threadfence();
    __syncthreads();
    if (tid == 0) {
      atomicAdd(bar, 1u);
      const unsigned int target = (unsigned int)(t + 1) * RNN_WGS;
      while (__hip_atomic_load(bar, __ATOMIC_RELAXED, __HIP_MEMORY_SCOPE_AGENT) < target)
        __builtin_amdgcn_s_sleep(2);
    }
    __syncthreads();
    __threadfence();
  }
}

// ---------------------------------------------------------------------------
// Final linear: Out[M x 51] = H[M x 512] @ Wlin[64(pad) x 512]^T + b_lin
// Pipelined loop fully unrolled (compile-time 16 k-steps -> pure renaming).
// ---------------------------------------------------------------------------
__global__ __launch_bounds__(256) void k_final_gemm(
    const unsigned short* __restrict__ A, const unsigned short* __restrict__ W,
    const float* __restrict__ bias, float* __restrict__ Out) {
  const int lane = threadIdx.x & 31;
  const int wave = threadIdx.x >> 5;
  const long m0 = (long)blockIdx.x * 64 + (wave >> 1) * 16;
  const int  n0 = (wave & 1) * 32;

  const unsigned short* pA = A + m0 * HD;
  const unsigned short* pW[2];
#pragma unroll
  for (int i = 0; i < 2; ++i) pW[i] = W + (long)(n0 + i * 16) * HD;

  f32x8 acc[2] = {};
  bf16x16 a = load_a(pA, HD, lane);
  bf16x16 b[2];
#pragma unroll
  for (int i = 0; i < 2; ++i) b[i] = load_b(pW[i], HD, lane);

#pragma unroll
  for (int kt = 1; kt < 16; ++kt) {
    const int k0 = kt << 5;
    bf16x16 an = load_a(pA + k0, HD, lane);
    bf16x16 bn[2];
#pragma unroll
    for (int i = 0; i < 2; ++i) bn[i] = load_b(pW[i] + k0, HD, lane);
#pragma unroll
    for (int i = 0; i < 2; ++i) acc[i] = wmma_bf16(a, b[i], acc[i]);
    a = an;
#pragma unroll
    for (int i = 0; i < 2; ++i) b[i] = bn[i];
  }
#pragma unroll
  for (int i = 0; i < 2; ++i) acc[i] = wmma_bf16(a, b[i], acc[i]);

  const int half = lane >> 4, ccol = lane & 15;
#pragma unroll
  for (int i = 0; i < 2; ++i) {
    const int col = n0 + i * 16 + ccol;
    if (col < FEAT) {
      const float bv = bias[col];
#pragma unroll
      for (int r = 0; r < 8; ++r)
        Out[(m0 + r + 8 * half) * (long)FEAT + col] = acc[i][r] + bv;
    }
  }
}

// ---------------------------------------------------------------------------
extern "C" void kernel_launch(void* const* d_in, const int* in_sizes, int n_in,
                              void* d_out, int out_size, void* d_ws, size_t ws_size,
                              hipStream_t stream) {
  (void)in_sizes; (void)n_in; (void)out_size; (void)ws_size;

  const float* x    = (const float*)d_in[0];
  const float* w    = (const float*)d_in[1];
  const float* Wih[3] = { (const float*)d_in[2], (const float*)d_in[6], (const float*)d_in[10] };
  const float* Whh[3] = { (const float*)d_in[3], (const float*)d_in[7], (const float*)d_in[11] };
  const float* bih[3] = { (const float*)d_in[4], (const float*)d_in[8], (const float*)d_in[12] };
  const float* bhh[3] = { (const float*)d_in[5], (const float*)d_in[9], (const float*)d_in[13] };
  const float* Wlin = (const float*)d_in[14];
  const float* blin = (const float*)d_in[15];

  char* ws = (char*)d_ws;
  size_t off = 0;
  auto alloc = [&](size_t bytes) -> void* {
    void* pp = ws + off;
    off += (bytes + 255) & ~(size_t)255;
    return pp;
  };

  float*          XP   = (float*)alloc((size_t)MROWS * G4 * 4);     // 512 MB
  unsigned short* HBa  = (unsigned short*)alloc((size_t)MROWS * HD * 2); // 64 MB
  unsigned short* HBb  = (unsigned short*)alloc((size_t)MROWS * HD * 2); // 64 MB
  unsigned short* Z0   = (unsigned short*)alloc((size_t)MROWS * K1 * 2); // 16 MB
  unsigned short* WIH1 = (unsigned short*)alloc((size_t)G4 * K1 * 2);
  unsigned short* WIH2 = (unsigned short*)alloc((size_t)G4 * HD * 2);
  unsigned short* WIH3 = (unsigned short*)alloc((size_t)G4 * HD * 2);
  unsigned short* WHHp[3] = {
    (unsigned short*)alloc((size_t)G4 * HD * 2),
    (unsigned short*)alloc((size_t)G4 * HD * 2),
    (unsigned short*)alloc((size_t)G4 * HD * 2) };
  unsigned short* WLIN = (unsigned short*)alloc((size_t)64 * HD * 2);
  float* BIAS[3] = { (float*)alloc(G4 * 4), (float*)alloc(G4 * 4), (float*)alloc(G4 * 4) };
  unsigned short* H0   = (unsigned short*)alloc((size_t)NB * HD * 2);
  unsigned int*   BAR  = (unsigned int*)alloc(256);

  // ---- packing / init (runs every call; ws is poisoned by the harness) ----
  hipMemsetAsync(H0, 0, (size_t)NB * HD * 2, stream);

  {
    long tot = (long)MROWS * K1;
    k_pack_z0<<<(unsigned)(tot / 256), 256, 0, stream>>>(x, w, Z0);
  }
  k_pack_mat<<<(G4 * K1) / 256, 256, 0, stream>>>(Wih[0], WIH1, G4, 2 * FEAT, K1, (long)G4 * K1);
  k_pack_mat<<<(G4 * HD) / 256, 256, 0, stream>>>(Wih[1], WIH2, G4, HD, HD, (long)G4 * HD);
  k_pack_mat<<<(G4 * HD) / 256, 256, 0, stream>>>(Wih[2], WIH3, G4, HD, HD, (long)G4 * HD);
  for (int l = 0; l < 3; ++l)
    k_pack_mat<<<(G4 * HD) / 256, 256, 0, stream>>>(Whh[l], WHHp[l], G4, HD, HD, (long)G4 * HD);
  k_pack_mat<<<(64 * HD) / 256, 256, 0, stream>>>(Wlin, WLIN, FEAT, HD, HD, (long)64 * HD);
  for (int l = 0; l < 3; ++l)
    k_bias_add<<<G4 / 256, 256, 0, stream>>>(bih[l], bhh[l], BIAS[l], G4);

  const unsigned short* wihp[3] = { WIH1, WIH2, WIH3 };
  const int Ks[3] = { K1, HD, HD };
  const unsigned short* layer_in[3]  = { Z0, HBa, HBb };
  unsigned short*       layer_out[3] = { HBa, HBb, HBa };

  for (int l = 0; l < 3; ++l) {
    // x_proj = z @ Wih^T + (b_ih + b_hh)   (full-grid WMMA GEMM)
    k_proj_gemm<<<dim3(G4 / 128, MROWS / 64), 256, 0, stream>>>(
        layer_in[l], wihp[l], BIAS[l], XP, Ks[l]);
    // sequential recurrence (persistent kernel, 8 WGs, 1024 grid barriers)
    hipMemsetAsync(BAR, 0, sizeof(unsigned int), stream);
    k_lstm_rec<<<RNN_WGS, 256, 0, stream>>>(WHHp[l], XP, H0, layer_out[l], BAR);
  }

  k_final_gemm<<<MROWS / 64, 256, 0, stream>>>(layer_out[2], WLIN, blin, (float*)d_out);
}
